// cosFormerAttention_73409581023761
// MI455X (gfx1250) — matrix operand order repaired and checked by reading, not compiled, not run on hardware
//
#include <hip/hip_runtime.h>
#include <hip/hip_bf16.h>
#include <math.h>

// ---------------------------------------------------------------------------
// cosFormer linear attention for MI455X (gfx1250, wave32, WMMA)
// B=4 H=16 S=4096 D=64, f32 in/out, bf16 WMMA with f32 accumulation.
// LDS tiles are staged in WMMA fragment order (k-permuted for A, [N][K] for B)
// with an XOR 32B-block swizzle, so fragments load as 2x ds_load_b128.
// ---------------------------------------------------------------------------

typedef __bf16 bf16_t;
typedef bf16_t v16bf __attribute__((ext_vector_type(16)));
typedef bf16_t v8bf  __attribute__((ext_vector_type(8)));
typedef bf16_t v4bf  __attribute__((ext_vector_type(4)));
typedef float  v8f   __attribute__((ext_vector_type(8)));

#define B_ 4
#define H_ 16
#define S_ 4096
#define D_ 64
#define BH_ (B_ * H_)

#define CHUNK 128   // S rows per workgroup, pass 1
#define SROW  128   // S rows per workgroup, pass 2

// workspace layout (float elements)
#define KV_ELEMS (BH_ * D_ * D_)
#define WS_KVSIN 0
#define WS_KVCOS (KV_ELEMS)
#define WS_KSSIN (2 * KV_ELEMS)
#define WS_KSCOS (2 * KV_ELEMS + BH_ * D_)
#define WS_TOTAL (2 * KV_ELEMS + 2 * BH_ * D_)

#define PI_HALF 1.57079632679489662f

// A-operand k-permutation within a 32-k group (involution: swap bits 3<->4).
// Positions 0..15 hold ks {0..7,16..23} (low lanes), 16..31 hold {8..15,24..31}.
__device__ __forceinline__ int swap34(int x) {
    return (x & ~24) | ((x & 8) << 1) | ((x & 16) >> 1);
}

// XOR swizzle of 16-element (32B) blocks within a row; key spreads banks for
// both row-step-4 staging stores and row-step-1 fragment reads.
__device__ __forceinline__ int swz(int row, int pos, int rowlen, int keymask) {
    const int blk = pos >> 4;
    const int key = ((row >> 2) ^ row) & keymask;
    return row * rowlen + (((blk ^ key) << 4) | (pos & 15));
}

__global__ void zero_ws_kernel(float* __restrict__ ws, int n) {
    int i = blockIdx.x * blockDim.x + threadIdx.x;
    if (i < n) ws[i] = 0.0f;
}

// ---------------------------------------------------------------------------
// Pass 1: per-head KV moment matrices + key sums.
//   kv_x[d][e] += sum_s x(s)*relu(K[s][d])*V[s][e],  ksum_x[d] += sum_s x(s)*relu(K[s][d])
// grid: BH_ * (S_/CHUNK) blocks of 256 threads (8 waves).
// Waves 0-3 -> sin matrix (16-row d-strips), waves 4-7 -> cos matrix.
// ---------------------------------------------------------------------------
__global__ __launch_bounds__(256)
void cosformer_kv_kernel(const float* __restrict__ Kg,
                         const float* __restrict__ Vg,
                         const unsigned char* __restrict__ maskg,
                         float* __restrict__ ws) {
    __shared__ alignas(32) bf16_t lds_ksin[D_ * CHUNK];  // A: [d][s_perm, swz]
    __shared__ alignas(32) bf16_t lds_kcos[D_ * CHUNK];  // A: [d][s_perm, swz]
    __shared__ alignas(32) bf16_t lds_vv [D_ * CHUNK];   // B: [e][s_perm, swz]
    __shared__ float row_sin[CHUNK];
    __shared__ float row_cos[CHUNK];

    const int tid  = threadIdx.x;
    const int bh   = blockIdx.x / (S_ / CHUNK);
    const int chnk = blockIdx.x % (S_ / CHUNK);
    const int b    = bh / H_;
    const int s0   = chnk * CHUNK;

    // per-row positional weights with mask folded in
    if (tid < CHUNK) {
        const int s = s0 + tid;
        const float idx = (float)(s + 1) * (1.0f / (float)S_);
        const float m = maskg[(size_t)b * S_ + s] ? 1.0f : 0.0f;
        row_sin[tid] = __sinf(idx * PI_HALF) * m;
        row_cos[tid] = __cosf(idx * PI_HALF) * m;
    }
    __syncthreads();

    // stage K (relu + sin/cos scale) and V as bf16, transposed into
    // fragment-ready layouts
    const float4* K4 = (const float4*)(Kg + ((size_t)bh * S_ + s0) * D_);
    const float4* V4 = (const float4*)(Vg + ((size_t)bh * S_ + s0) * D_);
#pragma unroll
    for (int j = 0; j < 8; ++j) {
        const int v4i = tid + j * 256;
        const int r = v4i >> 4;          // s row
        const int d = (v4i & 15) * 4;    // first of 4 d (or e) indices
        const float4 kq = K4[v4i];
        const float4 vq = V4[v4i];
        const float sn = row_sin[r];
        const float cs = row_cos[r];
        const float k[4] = {fmaxf(kq.x, 0.f), fmaxf(kq.y, 0.f),
                            fmaxf(kq.z, 0.f), fmaxf(kq.w, 0.f)};
        const float v[4] = {vq.x, vq.y, vq.z, vq.w};
        const int sp = (r & 96) | swap34(r & 31);   // A k-permuted s position
#pragma unroll
        for (int t = 0; t < 4; ++t) {
            lds_ksin[swz(d + t, sp, CHUNK, 7)] = (bf16_t)(k[t] * sn);
            lds_kcos[swz(d + t, sp, CHUNK, 7)] = (bf16_t)(k[t] * cs);
            lds_vv [swz(d + t, sp, CHUNK, 7)]  = (bf16_t)v[t];
        }
    }
    __syncthreads();

    // key-sum partial reduction (waves 0-3 fully active; permutation/swizzle
    // within a row is sum-invariant, so read linearly with wide loads)
    if (tid < 2 * D_) {
        const bf16_t* src = (tid < D_) ? lds_ksin : lds_kcos;
        const int d = tid & (D_ - 1);
        float acc = 0.0f;
#pragma unroll
        for (int j = 0; j < CHUNK; j += 8) {
            const v8bf v = *(const v8bf*)&src[d * CHUNK + j];
#pragma unroll
            for (int t = 0; t < 8; ++t) acc += (float)v[t];
        }
        atomicAdd(ws + ((tid < D_) ? WS_KSSIN : WS_KSCOS) + bh * D_ + d, acc);
    }

    // WMMA: kv_x[64x64] += A(16x32 strips of K_x^T) @ B(V 32x16 tiles)
    const int wave = tid >> 5;
    const int lane = tid & 31;
    const int mat  = wave >> 2;              // 0 = sin, 1 = cos
    const int m_base = (wave & 3) * 16;      // d-strip
    const int lo = lane & 15;
    const int half16 = (lane & 16) ? 16 : 0; // lane-half block offset (A and B)
    const bf16_t* Asrc = mat ? lds_kcos : lds_ksin;
    float* kvdst = ws + (mat ? WS_KVCOS : WS_KVSIN) + bh * (D_ * D_);

    v16bf afrag[4];   // hoisted: reused across all 4 N-tiles
#pragma unroll
    for (int kbi = 0; kbi < 4; ++kbi)
        afrag[kbi] = *(const v16bf*)&Asrc[swz(m_base + lo, kbi * 32 + half16, CHUNK, 7)];

    for (int nt = 0; nt < 4; ++nt) {
        v8f acc = {};
#pragma unroll
        for (int kbi = 0; kbi < 4; ++kbi) {
            const v16bf bb = *(const v16bf*)
                &lds_vv[swz(nt * 16 + lo, kbi * 32 + half16, CHUNK, 7)];
            acc = __builtin_amdgcn_wmma_f32_16x16x32_bf16(
                false, afrag[kbi], false, bb, (short)0, acc, false, false);
        }
#pragma unroll
        for (int i = 0; i < 8; ++i) {
            const int m = m_base + i + ((lane & 16) ? 8 : 0);
            atomicAdd(&kvdst[m * D_ + nt * 16 + lo], acc[i]);
        }
    }
}

// ---------------------------------------------------------------------------
// Pass 2: out[s][e] = (q_sin[s,:]@kv_sin + q_cos[s,:]@kv_cos)[e] / norm[s]
// grid: BH_ * (S_/SROW) blocks of 256 threads; wave w owns 16 Q rows.
// ---------------------------------------------------------------------------
__global__ __launch_bounds__(256)
void cosformer_out_kernel(const float* __restrict__ Qg,
                          const float* __restrict__ ws,
                          float* __restrict__ out) {
    __shared__ alignas(32) bf16_t lds_qsin[SROW * D_];  // A: [s][d_perm, swz]
    __shared__ alignas(32) bf16_t lds_qcos[SROW * D_];  // A: [s][d_perm, swz]
    __shared__ alignas(32) bf16_t lds_kvs[D_ * D_];     // B: [e][d_perm, swz]
    __shared__ alignas(32) bf16_t lds_kvc[D_ * D_];     // B: [e][d_perm, swz]
    __shared__ float lds_kss[D_];   // stored in A's d-permuted order
    __shared__ float lds_ksc[D_];
    __shared__ float row_sin[SROW];
    __shared__ float row_cos[SROW];
    __shared__ float lds_norm[SROW];

    const int tid  = threadIdx.x;
    const int bh   = blockIdx.x / (S_ / SROW);
    const int tile = blockIdx.x % (S_ / SROW);
    const int s0   = tile * SROW;

    if (tid < SROW) {
        const int s = s0 + tid;
        const float idx = (float)(s + 1) * (1.0f / (float)S_);
        row_sin[tid] = __sinf(idx * PI_HALF);
        row_cos[tid] = __cosf(idx * PI_HALF);
    }
    if (tid < 2 * D_) {
        const int d = tid & (D_ - 1);
        const int pd = (d & 32) | swap34(d & 31);   // match A's k-permutation
        if (tid < D_) lds_kss[pd] = ws[WS_KSSIN + bh * D_ + d];
        else          lds_ksc[pd] = ws[WS_KSCOS + bh * D_ + d];
    }
    __syncthreads();

    // stage Q tile (relu + sin/cos scale) as bf16 in A fragment order.
    // pd(d..d+3) is contiguous for d % 4 == 0, so store as one v4bf.
    const float4* Q4 = (const float4*)(Qg + ((size_t)bh * S_ + s0) * D_);
#pragma unroll
    for (int j = 0; j < 8; ++j) {
        const int v4i = tid + j * 256;
        const int r = v4i >> 4;
        const int d = (v4i & 15) * 4;
        const float4 q = Q4[v4i];
        const float sn = row_sin[r];
        const float cs = row_cos[r];
        const float q0 = fmaxf(q.x, 0.f), q1 = fmaxf(q.y, 0.f);
        const float q2 = fmaxf(q.z, 0.f), q3 = fmaxf(q.w, 0.f);
        const int pd = (d & 32) | swap34(d & 31);
        const int base = swz(r, pd, D_, 3);
        const v4bf vs = {(bf16_t)(q0 * sn), (bf16_t)(q1 * sn),
                         (bf16_t)(q2 * sn), (bf16_t)(q3 * sn)};
        *(v4bf*)&lds_qsin[base] = vs;
        const v4bf vc = {(bf16_t)(q0 * cs), (bf16_t)(q1 * cs),
                         (bf16_t)(q2 * cs), (bf16_t)(q3 * cs)};
        *(v4bf*)&lds_qcos[base] = vc;
    }

    // stage the head's KV matrices as bf16, transposed into B layout [e][d]
    // (kv fits in L2; lanes read consecutive e for coalescing)
    {
        const float* kvsg = ws + WS_KVSIN + bh * (D_ * D_);
        const float* kvcg = ws + WS_KVCOS + bh * (D_ * D_);
        const int e  = tid & (D_ - 1);
        const int dg = (tid >> 6) * 16;
#pragma unroll
        for (int j = 0; j < 16; ++j) {
            const int d = dg + j;
            lds_kvs[swz(e, (d & 32) | swap34(d & 31), D_, 3)] = (bf16_t)kvsg[d * D_ + e];
            lds_kvc[swz(e, (d & 32) | swap34(d & 31), D_, 3)] = (bf16_t)kvcg[d * D_ + e];
        }
    }
    __syncthreads();

    // normalizer: norm[s] = q_sin[s,:].ksum_sin + q_cos[s,:].ksum_cos
    // (walk logical 16-elem blocks through the same swizzle; ksum arrays are
    //  already in the matching d-permuted order)
    if (tid < SROW) {
        const int key = ((tid >> 2) ^ tid) & 3;
        float acc = 0.0f;
#pragma unroll
        for (int blk = 0; blk < 4; ++blk) {
            const int sb = tid * D_ + ((blk ^ key) << 4);
            const v8bf a0 = *(const v8bf*)&lds_qsin[sb];
            const v8bf a1 = *(const v8bf*)&lds_qsin[sb + 8];
            const v8bf c0 = *(const v8bf*)&lds_qcos[sb];
            const v8bf c1 = *(const v8bf*)&lds_qcos[sb + 8];
            const float* ks = &lds_kss[blk * 16];
            const float* kc = &lds_ksc[blk * 16];
#pragma unroll
            for (int t = 0; t < 8; ++t) {
                acc += (float)a0[t] * ks[t] + (float)a1[t] * ks[8 + t];
                acc += (float)c0[t] * kc[t] + (float)c1[t] * kc[8 + t];
            }
        }
        lds_norm[tid] = acc;
    }
    __syncthreads();

    const int wave = tid >> 5;
    const int lane = tid & 31;
    const int m_base = wave * 16;            // Q-row strip
    const int lo = lane & 15;
    const int half16 = (lane & 16) ? 16 : 0;
    float* outp = out + ((size_t)bh * S_ + s0) * D_;

    v16bf asf[2], acf[2];   // hoisted A fragments, reused across N-tiles
#pragma unroll
    for (int kbi = 0; kbi < 2; ++kbi) {
        const int aidx = swz(m_base + lo, kbi * 32 + half16, D_, 3);
        asf[kbi] = *(const v16bf*)&lds_qsin[aidx];
        acf[kbi] = *(const v16bf*)&lds_qcos[aidx];
    }

    float rn[8];            // per-lane reciprocal normalizers (rows fixed)
#pragma unroll
    for (int i = 0; i < 8; ++i)
        rn[i] = 1.0f / lds_norm[m_base + i + ((lane & 16) ? 8 : 0)];

    for (int nt = 0; nt < 4; ++nt) {
        v8f acc = {};
#pragma unroll
        for (int kbi = 0; kbi < 2; ++kbi) {
            const int bidx = swz(nt * 16 + lo, kbi * 32 + half16, D_, 3);
            const v16bf bs = *(const v16bf*)&lds_kvs[bidx];
            const v16bf bc = *(const v16bf*)&lds_kvc[bidx];
            acc = __builtin_amdgcn_wmma_f32_16x16x32_bf16(
                false, asf[kbi], false, bs, (short)0, acc, false, false);
            acc = __builtin_amdgcn_wmma_f32_16x16x32_bf16(
                false, acf[kbi], false, bc, (short)0, acc, false, false);
        }
#pragma unroll
        for (int i = 0; i < 8; ++i) {
            const int m = m_base + i + ((lane & 16) ? 8 : 0);
            outp[(size_t)m * D_ + nt * 16 + lo] = acc[i] * rn[i];
        }
    }
}

// ---------------------------------------------------------------------------
extern "C" void kernel_launch(void* const* d_in, const int* in_sizes, int n_in,
                              void* d_out, int out_size, void* d_ws, size_t ws_size,
                              hipStream_t stream) {
    (void)in_sizes; (void)n_in; (void)out_size; (void)ws_size;
    const float* Q = (const float*)d_in[0];
    const float* K = (const float*)d_in[1];
    const float* V = (const float*)d_in[2];
    const unsigned char* mask = (const unsigned char*)d_in[3];
    float* out = (float*)d_out;
    float* ws  = (float*)d_ws;

    zero_ws_kernel<<<(WS_TOTAL + 255) / 256, 256, 0, stream>>>(ws, WS_TOTAL);
    cosformer_kv_kernel<<<BH_ * (S_ / CHUNK), 256, 0, stream>>>(K, V, mask, ws);
    cosformer_out_kernel<<<BH_ * (S_ / SROW), 256, 0, stream>>>(Q, ws, out);
}
